// Local_Attention_35759897706549
// MI455X (gfx1250) — matrix operand order repaired and checked by reading, not compiled, not run on hardware
//
#include <hip/hip_runtime.h>

// ---------------------------------------------------------------------------
// Point-Transformer local attention for MI455X (gfx1250, wave32, WMMA)
// B=8, M=4096, K=16 neighbors, C=D=64
// ---------------------------------------------------------------------------

typedef __attribute__((ext_vector_type(16))) _Float16 v16h;
typedef __attribute__((ext_vector_type(8)))  _Float16 v8h;
typedef __attribute__((ext_vector_type(8)))  float    v8f;

#define WMMA(A, B, C) __builtin_amdgcn_wmma_f32_16x16x32_f16( \
    false, (A), false, (B), (short)0, (C), false, false)

static constexpr int KN    = 16;     // neighbors per token (tile M-dim)
static constexpr int CD    = 64;     // channels C == D
static constexpr int NTOK  = 8 * 4096;

// -------------------------------------------------------------------------
// B-operand LDS layout for V_WMMA_F32_16X16X32_F16 (32x16 f16 tile):
//   lane L holds column n = L%16; halves h=0..15 cover k = ck*32 + (L>=16?8:0) + (h&7) + (h>=8?16:0)
// Stored lane-major so each lane reads one contiguous 32B run (2x ds_load_b128).
// -------------------------------------------------------------------------
__device__ __forceinline__ const v16h* bptr(const _Float16* base, int ck, int nt, int lane) {
    return (const v16h*)(base + ((ck * 4 + nt) * 32 + lane) * 16);
}

// Build B-layout f16 copy of a 64x64 weight W (row-major [n][k], lin = t @ W^T)
__device__ __forceinline__ void build_B64(_Float16* dst, const float* __restrict__ W,
                                          int tid, int nthr) {
    for (int idx = tid; idx < 4096; idx += nthr) {
        int h  = idx & 15;
        int L  = (idx >> 4) & 31;
        int nt = (idx >> 9) & 3;
        int ck = idx >> 11;
        int n  = nt * 16 + (L & 15);
        int k  = ck * 32 + ((L >= 16) ? 8 : 0) + (h & 7) + ((h >= 8) ? 16 : 0);
        dst[idx] = (_Float16)W[n * 64 + k];
    }
}

// Wp1 is 64x3; pad K to 32 with zeros (single K-chunk).
__device__ __forceinline__ void build_Bp1(_Float16* dst, const float* __restrict__ W,
                                          int tid, int nthr) {
    for (int idx = tid; idx < 2048; idx += nthr) {
        int h  = idx & 15;
        int L  = (idx >> 4) & 31;
        int nt = idx >> 9;
        int n  = nt * 16 + (L & 15);
        int k  = ((L >= 16) ? 8 : 0) + (h & 7) + ((h >= 8) ? 16 : 0);
        dst[idx] = (k < 3) ? (_Float16)W[n * 3 + k] : (_Float16)0.0f;
    }
}

// -------------------------------------------------------------------------
// A-operand pack for 16x32 f16 tile:
//   lane L: row r = L%16, kb = (L>=16)*8; halves 0..7 <- k=kb..kb+7,
//   halves 8..15 <- k=kb+16..kb+23 (within the 32-wide K chunk).
// Source row pointer must already include  r*stride + ck*32 + kb.
// -------------------------------------------------------------------------
__device__ __forceinline__ v16h packA_f32(const float* __restrict__ p) {
    float4 f0 = *(const float4*)(p);
    float4 f1 = *(const float4*)(p + 4);
    float4 f2 = *(const float4*)(p + 16);
    float4 f3 = *(const float4*)(p + 20);
    v16h a;
    a[0] = (_Float16)f0.x;  a[1] = (_Float16)f0.y;  a[2]  = (_Float16)f0.z;  a[3]  = (_Float16)f0.w;
    a[4] = (_Float16)f1.x;  a[5] = (_Float16)f1.y;  a[6]  = (_Float16)f1.z;  a[7]  = (_Float16)f1.w;
    a[8] = (_Float16)f2.x;  a[9] = (_Float16)f2.y;  a[10] = (_Float16)f2.z;  a[11] = (_Float16)f2.w;
    a[12] = (_Float16)f3.x; a[13] = (_Float16)f3.y; a[14] = (_Float16)f3.z;  a[15] = (_Float16)f3.w;
    return a;
}

__device__ __forceinline__ v16h packA_lds(const _Float16* p) {
    v8h lo = *(const v8h*)(p);
    v8h hi = *(const v8h*)(p + 16);
    v16h a;
#pragma unroll
    for (int i = 0; i < 8; ++i) { a[i] = lo[i]; a[8 + i] = hi[i]; }
    return a;
}

// ===========================================================================
// Kernel 1: per-token attention. One wave32 = one token's 16x64 tile.
// Produces res[token][64] into workspace.
// ===========================================================================
__global__ __launch_bounds__(128) void local_attn_kernel(
    const float* __restrict__ x,   const float* __restrict__ xyz,
    const float* __restrict__ Wq,  const float* __restrict__ bq,
    const float* __restrict__ Wk,  const float* __restrict__ bk,
    const float* __restrict__ Wv,  const float* __restrict__ bv,
    const float* __restrict__ Wp1, const float* __restrict__ bp1,
    const float* __restrict__ Wp2, const float* __restrict__ bp2,
    const float* __restrict__ Wa1, const float* __restrict__ ba1,
    const float* __restrict__ Wa2, const float* __restrict__ ba2,
    float* __restrict__ res_ws)
{
    // LDS: 6 weights in B-layout (8KB each) + Wp1 (4KB) + per-wave scratch (2KB x 4)
    __shared__ _Float16 sW[6][4096];      // 0=Wq 1=Wk 2=Wv 3=Wp2 4=Wa1 5=Wa2
    __shared__ _Float16 sP1[2048];
    __shared__ _Float16 sScr[4][KN * CD]; // per-wave 16x64 f16 scratch

    const int tid = threadIdx.x;
    build_B64(sW[0], Wq,  tid, 128);
    build_B64(sW[1], Wk,  tid, 128);
    build_B64(sW[2], Wv,  tid, 128);
    build_B64(sW[3], Wp2, tid, 128);
    build_B64(sW[4], Wa1, tid, 128);
    build_B64(sW[5], Wa2, tid, 128);
    build_Bp1(sP1, Wp1, tid, 128);
    __syncthreads();

    const int lane = tid & 31;
    const int wv   = tid >> 5;
    const int r    = lane & 15;        // A-row / C-column index for this lane
    const int hi16 = lane >> 4;        // 0: rows 0..7 of C/D, 1: rows 8..15
    const int kb   = hi16 * 8;         // A-layout K sub-offset
    _Float16* scr  = sScr[wv];

    for (int it = 0; it < 8; ++it) {
        const int tok = (blockIdx.x * 4 + wv) * 8 + it;
        const float* xt = x + (size_t)tok * (KN * CD);
        if (it < 7) __builtin_prefetch(xt + KN * CD, 0, 0);  // next token's tile

        // ---- A tiles of x (f32 -> f16) -------------------------------------
        const float* rp = xt + r * CD + kb;
        v16h ax0 = packA_f32(rp);
        v16h ax1 = packA_f32(rp + 32);

        // ---- rel = xyz - xyz[0], K padded 3 -> 32 --------------------------
        v16h arel;
#pragma unroll
        for (int i = 0; i < 16; ++i) arel[i] = (_Float16)0.0f;
        {
            const float* xp = xyz + (size_t)tok * (KN * 3);
            if (lane < 16) {
                arel[0] = (_Float16)(xp[r * 3 + 0] - xp[0]);
                arel[1] = (_Float16)(xp[r * 3 + 1] - xp[1]);
                arel[2] = (_Float16)(xp[r * 3 + 2] - xp[2]);
            }
        }

        // ---- q row (only row 0 of x @ Wq^T is valid; shfl-broadcast it) ----
        float qv[4];
#pragma unroll
        for (int nt = 0; nt < 4; ++nt) {
            v8f c = {};
            c = WMMA(ax0, *bptr(sW[0], 0, nt, lane), c);
            c = WMMA(ax1, *bptr(sW[0], 1, nt, lane), c);
            float q0 = __shfl(c[0], r, 32);        // (m=0, n=r) lives in lanes 0..15
            qv[nt] = q0 + bq[nt * 16 + r];
        }

        // ---- p1 = relu(rel @ Wp1^T + bp1) -> scratch -----------------------
#pragma unroll
        for (int nt = 0; nt < 4; ++nt) {
            v8f c = {};
            c = WMMA(arel, *(const v16h*)(sP1 + (nt * 32 + lane) * 16), c);
            const float bb = bp1[nt * 16 + r];
#pragma unroll
            for (int i = 0; i < 8; ++i) {
                float t = c[i] + bb;
                t = t > 0.0f ? t : 0.0f;
                scr[(i + 8 * hi16) * CD + nt * 16 + r] = (_Float16)t;
            }
        }
        const _Float16* sp = scr + r * CD + kb;
        v16h ap0 = packA_lds(sp);
        v16h ap1 = packA_lds(sp + 32);

        // ---- pos_enc = p1 @ Wp2^T + bp2 (kept live in VGPRs) ---------------
        v8f pe[4];
#pragma unroll
        for (int nt = 0; nt < 4; ++nt) {
            v8f c = {};
            c = WMMA(ap0, *bptr(sW[3], 0, nt, lane), c);
            c = WMMA(ap1, *bptr(sW[3], 1, nt, lane), c);
            const float bb = bp2[nt * 16 + r];
#pragma unroll
            for (int i = 0; i < 8; ++i) c[i] += bb;
            pe[nt] = c;
        }

        // ---- h = q - (x @ Wk^T + bk) + pos_enc -> scratch ------------------
#pragma unroll
        for (int nt = 0; nt < 4; ++nt) {
            v8f c = {};
            c = WMMA(ax0, *bptr(sW[1], 0, nt, lane), c);
            c = WMMA(ax1, *bptr(sW[1], 1, nt, lane), c);
            const float hb = qv[nt] - bk[nt * 16 + r];
#pragma unroll
            for (int i = 0; i < 8; ++i)
                scr[(i + 8 * hi16) * CD + nt * 16 + r] = (_Float16)(hb - c[i] + pe[nt][i]);
        }
        v16h ah0 = packA_lds(sp);
        v16h ah1 = packA_lds(sp + 32);

        // ---- t1 = relu(h @ Wa1^T + ba1) -> scratch -------------------------
#pragma unroll
        for (int nt = 0; nt < 4; ++nt) {
            v8f c = {};
            c = WMMA(ah0, *bptr(sW[4], 0, nt, lane), c);
            c = WMMA(ah1, *bptr(sW[4], 1, nt, lane), c);
            const float bb = ba1[nt * 16 + r];
#pragma unroll
            for (int i = 0; i < 8; ++i) {
                float t = c[i] + bb;
                t = t > 0.0f ? t : 0.0f;
                scr[(i + 8 * hi16) * CD + nt * 16 + r] = (_Float16)t;
            }
        }
        v16h at0 = packA_lds(sp);
        v16h at1 = packA_lds(sp + 32);

        // ---- logits, v, softmax over the 16 neighbors, weighted sum --------
#pragma unroll
        for (int nt = 0; nt < 4; ++nt) {
            v8f lg = {};
            lg = WMMA(at0, *bptr(sW[5], 0, nt, lane), lg);
            lg = WMMA(at1, *bptr(sW[5], 1, nt, lane), lg);
            v8f vv = {};
            vv = WMMA(ax0, *bptr(sW[2], 0, nt, lane), vv);
            vv = WMMA(ax1, *bptr(sW[2], 1, nt, lane), vv);

            const float ba2v = ba2[nt * 16 + r];
            const float bvv  = bv[nt * 16 + r];

            float lgf[8];
            float mx = -3.0e30f;
#pragma unroll
            for (int i = 0; i < 8; ++i) {
                lgf[i] = lg[i] + ba2v;
                mx = fmaxf(mx, lgf[i]);
            }
            // column spans lane pair (L, L^16): finish reductions across it
            mx = fmaxf(mx, __shfl_xor(mx, 16, 32));

            float s = 0.0f, acc = 0.0f;
#pragma unroll
            for (int i = 0; i < 8; ++i) {
                const float e = __expf((lgf[i] - mx) * 0.125f);  // 1/sqrt(64)
                s   += e;
                acc += e * (vv[i] + bvv + pe[nt][i]);
            }
            s   += __shfl_xor(s, 16, 32);
            acc += __shfl_xor(acc, 16, 32);

            if (lane < 16)
                res_ws[(size_t)tok * CD + nt * 16 + r] = acc / s;
        }
    }
}

// ===========================================================================
// Kernel 2: out = res @ W2^T + b2 + raw_feature. One wave = 16 tokens.
// ===========================================================================
__global__ __launch_bounds__(256) void out_proj_kernel(
    const float* __restrict__ res_ws, const float* __restrict__ W2,
    const float* __restrict__ b2,     const float* __restrict__ x,
    float* __restrict__ out)
{
    __shared__ _Float16 sW2[4096];
    build_B64(sW2, W2, threadIdx.x, 256);
    __syncthreads();

    const int lane = threadIdx.x & 31;
    const int wv   = threadIdx.x >> 5;
    const int r    = lane & 15;
    const int hi16 = lane >> 4;
    const int kb   = hi16 * 8;
    const int tb   = (blockIdx.x * 8 + wv) * 16;   // 16 tokens per wave

    const float* rp = res_ws + (size_t)(tb + r) * CD + kb;
    v16h a0 = packA_f32(rp);
    v16h a1 = packA_f32(rp + 32);

#pragma unroll
    for (int nt = 0; nt < 4; ++nt) {
        v8f c = {};
        c = WMMA(a0, *bptr(sW2, 0, nt, lane), c);
        c = WMMA(a1, *bptr(sW2, 1, nt, lane), c);
        const int   n  = nt * 16 + r;
        const float bb = b2[n];
#pragma unroll
        for (int i = 0; i < 8; ++i) {
            const int tok = tb + i + 8 * hi16;
            out[(size_t)tok * CD + n] =
                c[i] + bb + x[(size_t)tok * (KN * CD) + n];  // + raw_feature
        }
    }
}

// ===========================================================================
extern "C" void kernel_launch(void* const* d_in, const int* in_sizes, int n_in,
                              void* d_out, int out_size, void* d_ws, size_t ws_size,
                              hipStream_t stream) {
    (void)in_sizes; (void)n_in; (void)out_size; (void)ws_size;
    const float* x   = (const float*)d_in[0];
    const float* xyz = (const float*)d_in[1];
    const float* Wq  = (const float*)d_in[2];
    const float* bq  = (const float*)d_in[3];
    const float* Wk  = (const float*)d_in[4];
    const float* bk  = (const float*)d_in[5];
    const float* Wv  = (const float*)d_in[6];
    const float* bv  = (const float*)d_in[7];
    const float* Wp1 = (const float*)d_in[8];
    const float* bp1 = (const float*)d_in[9];
    const float* Wp2 = (const float*)d_in[10];
    const float* bp2 = (const float*)d_in[11];
    const float* Wa1 = (const float*)d_in[12];
    const float* ba1 = (const float*)d_in[13];
    const float* Wa2 = (const float*)d_in[14];
    const float* ba2 = (const float*)d_in[15];
    const float* W2  = (const float*)d_in[16];
    const float* b2  = (const float*)d_in[17];

    float* res_ws = (float*)d_ws;   // NTOK x 64 f32 = 8 MB
    float* out    = (float*)d_out;

    // 1024 blocks x 4 waves x 8 tokens = 32768 tokens
    local_attn_kernel<<<1024, 128, 0, stream>>>(
        x, xyz, Wq, bq, Wk, bk, Wv, bv, Wp1, bp1, Wp2, bp2,
        Wa1, ba1, Wa2, ba2, res_ws);

    // 256 blocks x 8 waves x 16 tokens = 32768 tokens
    out_proj_kernel<<<256, 256, 0, stream>>>(res_ws, W2, b2, x, out);
}
